// Ra_GC_20564303414150
// MI455X (gfx1250) — matrix operand order repaired and checked by reading, not compiled, not run on hardware
//
#include <hip/hip_runtime.h>
#include <hip/hip_bf16.h>
#include <stdint.h>

#define NODE    2048
#define CLUSTER 32
#define IN_DIM  64
#define OUT_DIM 64
#define BATCH   32

typedef __attribute__((ext_vector_type(16))) __bf16 v16bf;
typedef __attribute__((ext_vector_type(8)))  float  v8f;
typedef __attribute__((ext_vector_type(2)))  float  vf2;   // native vector for NT loads

struct U8x32 { uint4 lo; uint4 hi; };

static __device__ __forceinline__ unsigned int pack_bf16x2(float f0, float f1) {
    unsigned short a = __builtin_bit_cast(unsigned short, (__bf16)f0);
    unsigned short b = __builtin_bit_cast(unsigned short, (__bf16)f1);
    return (unsigned int)a | ((unsigned int)b << 16);
}

static __device__ __forceinline__ void wait_asynccnt0() {
#if __has_builtin(__builtin_amdgcn_s_wait_asynccnt)
    __builtin_amdgcn_s_wait_asynccnt(0);
#else
    asm volatile("s_wait_asynccnt 0x0" ::: "memory");
#endif
}

// Copy 16B global -> LDS via the CDNA5 async data path (ASYNCcnt-tracked).
static __device__ __forceinline__ void async_copy_b128(unsigned lds_byte_off,
                                                       const unsigned* gsrc) {
    asm volatile("global_load_async_to_lds_b128 %0, %1, off"
                 :: "v"(lds_byte_off), "v"((unsigned long long)(size_t)gsrc)
                 : "memory");
}

// ---------------------------------------------------------------------------
// Stage 1: per-node hypernetwork weight generation + small matmul (VALU, tiny)
//   w[n] = alpha[n,:] @ beta (64x64, LDS); h2[b,n,:] = h[b,n,:] @ w[n] + bias
// ---------------------------------------------------------------------------
__global__ __launch_bounds__(256) void hypernet_h2_kernel(
    const float* __restrict__ h, const float* __restrict__ alpha,
    const float* __restrict__ beta, const float* __restrict__ bias,
    float* __restrict__ h2)
{
    __shared__ float s_alpha[CLUSTER];
    __shared__ float s_w[IN_DIM * OUT_DIM];
    __shared__ float s_h[BATCH * IN_DIM];

    const int n = blockIdx.x;
    const int t = threadIdx.x;

    if (t < CLUSTER) s_alpha[t] = alpha[n * CLUSTER + t];
#pragma unroll
    for (int r = 0; r < 8; ++r) {
        int idx = t + 256 * r;
        int b = idx >> 6, i = idx & 63;
        s_h[idx] = h[((size_t)b * NODE + n) * IN_DIM + i];
    }
    __syncthreads();

#pragma unroll
    for (int r = 0; r < 16; ++r) {
        int idx = t + 256 * r;              // idx = i*64 + o
        float acc = 0.f;
#pragma unroll
        for (int c = 0; c < CLUSTER; ++c)
            acc = fmaf(s_alpha[c], beta[c * (IN_DIM * OUT_DIM) + idx], acc);
        s_w[idx] = acc;
    }
    __syncthreads();

#pragma unroll
    for (int r = 0; r < 8; ++r) {
        int idx = t + 256 * r;
        int b = idx >> 6, o = idx & 63;
        float acc = bias[o];
#pragma unroll
        for (int i = 0; i < IN_DIM; ++i)
            acc = fmaf(s_h[b * IN_DIM + i], s_w[i * OUT_DIM + o], acc);
        h2[((size_t)b * NODE + n) * OUT_DIM + o] = acc;
    }
}

// ---------------------------------------------------------------------------
// Stage 1b: pack/transpose h2 f32 -> h2sw bf16 pairs in the WMMA-B swizzle:
//   h2sw[b][o][npair] : u32 = { bf16(h2[b][2*npair][o]), bf16(h2[b][2*npair+1][o]) }
// So stage-2 B staging becomes a contiguous 16B-granular async copy.
// Grid (NODE/64, BATCH), block 256, LDS-tiled transpose, coalesced both sides.
// ---------------------------------------------------------------------------
__global__ __launch_bounds__(256) void h2_swizzle_kernel(
    const float* __restrict__ h2, unsigned int* __restrict__ h2sw)
{
    __shared__ float tile[64][65];
    const int b  = blockIdx.y;
    const int n0 = blockIdx.x * 64;
    const int t  = threadIdx.x;

    const float* src = h2 + ((size_t)b * NODE + n0) * OUT_DIM;
#pragma unroll
    for (int r = 0; r < 16; ++r) {          // 4096 f32, o-fastest (coalesced)
        int idx = t + 256 * r;
        int n = idx >> 6, o = idx & 63;
        tile[n][o] = src[n * OUT_DIM + o];
    }
    __syncthreads();

#pragma unroll
    for (int r = 0; r < 8; ++r) {           // 2048 u32, npair-fastest (coalesced)
        int g  = t + 256 * r;
        int o  = g >> 5;                    // 0..63
        int np = g & 31;                    // 0..31
        h2sw[((size_t)b * OUT_DIM + o) * (NODE / 2) + (n0 >> 1) + np] =
            pack_bf16x2(tile[2 * np][o], tile[2 * np + 1][o]);
    }
}

// ---------------------------------------------------------------------------
// Stage 2: out[b] = a[b] @ h2[b] + h2[b]   (bf16 WMMA, f32 accumulate)
// 8 waves/block, wave => 16x64 strip (4x v_wmma_f32_16x16x32_bf16 accumulators).
// B chunks arrive via double-buffered GLOBAL_LOAD_ASYNC_TO_LDS_B128 from the
// pre-swizzled h2sw; A streamed nontemporal from global + cvt to bf16.
// ---------------------------------------------------------------------------
#define BK        64
#define LDS_PITCH 36   // 32 kpairs + pad; 144B row (16B aligned, conflict-light)

static __device__ __forceinline__ void issue_b_stage(
    unsigned int* s_buf, const unsigned int* h2swb, int kc, int t)
{
#pragma unroll
    for (int r = 0; r < 2; ++r) {           // 512 x 16B = 8 KiB chunk
        int g = t + 256 * r;
        int o = g >> 3;                     // 0..63
        int q = (g & 7) << 2;               // kpair base 0,4,...,28
        unsigned lds = (unsigned)(size_t)&s_buf[o * LDS_PITCH + q];
        const unsigned* gsrc = h2swb + (size_t)o * (NODE / 2) + (kc >> 1) + q;
        async_copy_b128(lds, gsrc);
    }
}

__global__ __launch_bounds__(256) void mix_wmma_kernel(
    const float* __restrict__ a, const float* __restrict__ h2,
    const unsigned int* __restrict__ h2sw, float* __restrict__ out)
{
    __shared__ unsigned int s_b[2][OUT_DIM * LDS_PITCH];

    const int t    = threadIdx.x;
    const int lane = t & 31;
    const int wv   = t >> 5;
    const int lm   = lane & 15;   // M index for A / N index within tile for B,C
    const int lh   = lane >> 4;   // lane-half selector
    const int b    = blockIdx.y;

    const int row0 = blockIdx.x * 128 + wv * 16 + lm;   // this lane's A row (M)
    const float*        __restrict__ arow  = a + ((size_t)b * NODE + row0) * (size_t)NODE;
    const unsigned int* __restrict__ h2swb = h2sw + (size_t)b * OUT_DIM * (NODE / 2);

    v8f acc0 = {}, acc1 = {}, acc2 = {}, acc3 = {};

    issue_b_stage(s_b[0], h2swb, 0, t);     // prologue prefetch

    for (int kc = 0; kc < NODE; kc += BK) {
        const int cur = (kc >> 6) & 1;
        wait_asynccnt0();                   // buffer `cur` landed in LDS
        __syncthreads();                    // visible to all waves; buf cur^1 free
        if (kc + BK < NODE)
            issue_b_stage(s_b[cur ^ 1], h2swb, kc + BK, t);  // overlap with compute

#pragma unroll
        for (int s = 0; s < 2; ++s) {       // two 16x16x32 K-steps per chunk
            // ---- A fragment: 16x32 bf16, contiguous k order (same as B) ----
            // lanes 0-15: K = {0..15}, lanes 16-31: K = {16..31}
            v16bf af;
#pragma unroll
            for (int v = 0; v < 8; ++v) {
                int k = kc + s * 32 + (v << 1) + (lh << 4);
                vf2 f2 = __builtin_nontemporal_load((const vf2*)(arow + k));
                af[2 * v]     = (__bf16)f2.x;
                af[2 * v + 1] = (__bf16)f2.y;
            }
            // ---- B fragments from LDS (two ds_load_b128 each) + WMMA ----
#pragma unroll
            for (int j = 0; j < 4; ++j) {
                int base = (j * 16 + lm) * LDS_PITCH + s * 16 + (lh << 3);
                U8x32 raw;
                raw.lo = *(const uint4*)&s_b[cur][base];
                raw.hi = *(const uint4*)&s_b[cur][base + 4];
                v16bf bf = __builtin_bit_cast(v16bf, raw);
                v8f* accp = (j == 0) ? &acc0 : (j == 1) ? &acc1 : (j == 2) ? &acc2 : &acc3;
                *accp = __builtin_amdgcn_wmma_f32_16x16x32_bf16(
                    false, af, false, bf, (short)0, *accp, false, false);
            }
        }
        __syncthreads();                    // done reading buf cur before its refill
    }

    // ---- epilogue: residual add + nontemporal store ----
    // C/D layout: VGPR r -> M=r (lanes 0-15), M=r+8 (lanes 16-31); N = lane%16
#pragma unroll
    for (int j = 0; j < 4; ++j) {
        v8f acc = (j == 0) ? acc0 : (j == 1) ? acc1 : (j == 2) ? acc2 : acc3;
        int o = j * 16 + lm;
#pragma unroll
        for (int r = 0; r < 8; ++r) {
            int m   = (lh << 3) + r;
            int row = blockIdx.x * 128 + wv * 16 + m;
            size_t off = ((size_t)b * NODE + row) * OUT_DIM + o;
            __builtin_nontemporal_store(acc[r] + h2[off], out + off);
        }
    }
}

// ---------------------------------------------------------------------------
extern "C" void kernel_launch(void* const* d_in, const int* in_sizes, int n_in,
                              void* d_out, int out_size, void* d_ws, size_t ws_size,
                              hipStream_t stream)
{
    const float* a     = (const float*)d_in[0];
    const float* h     = (const float*)d_in[1];
    const float* alpha = (const float*)d_in[2];
    const float* beta  = (const float*)d_in[3];
    const float* bias  = (const float*)d_in[4];
    float*       out   = (float*)d_out;

    float*        h2    = (float*)d_ws;                       // 16 MiB f32
    unsigned int* h2sw  = (unsigned int*)((char*)d_ws +
                          (size_t)BATCH * NODE * OUT_DIM * sizeof(float)); // +8 MiB

    hypernet_h2_kernel<<<NODE, 256, 0, stream>>>(h, alpha, beta, bias, h2);
    h2_swizzle_kernel<<<dim3(NODE / 64, BATCH), 256, 0, stream>>>(h2, h2sw);
    mix_wmma_kernel<<<dim3(NODE / 128, BATCH), 256, 0, stream>>>(a, h2, h2sw, out);
}
